// SparseConvBlock_5549097746523
// MI455X (gfx1250) — matrix operand order repaired and checked by reading, not compiled, not run on hardware
//
#include <hip/hip_runtime.h>

#define NPTS 131072
#define KOFF 27
#define CIN  64
#define COUT 128
#define TILE_M 128
#define NBLK (NPTS / TILE_M)      // 1024 workgroups
#define BN_EPS 1e-5f
#define NEG_SLOPE 0.01f

typedef __attribute__((ext_vector_type(16))) _Float16     v16h;
typedef __attribute__((ext_vector_type(8)))  _Float16     v8h;
typedef __attribute__((ext_vector_type(8)))  float        v8f;
typedef __attribute__((ext_vector_type(4)))  unsigned int v4u;

union Frag16 { v4u u[2]; v16h h; };

// ---- workspace layout (bytes) ----
#define WFRAG_ELEMS (KOFF * 2 * 8 * 32 * 16)        // 221184 f16 elements
#define OFF_WFRAG 0
#define OFF_BSUM  (WFRAG_ELEMS * 2)                 // 442368
#define OFF_BSQ   (OFF_BSUM + NBLK * COUT * 4)      // +524288
#define OFF_SCALE (OFF_BSQ  + NBLK * COUT * 4)
#define OFF_BIAS  (OFF_SCALE + COUT * 4)

// -------------------------------------------------------------------------
// Repack W[k][cin][cout] fp32 -> f16 B-fragment order:
// frag index = (((k*2 + kb)*8 + t)*32 + lane)*16 + e
// lane 0-15 : col = t*16+lane,    K = kb*32 + (e&7) + (e>>3)*16       (e = 0..15)
// lane 16-31: col = t*16+lane-16, K = kb*32 + 8 + (e&7) + (e>>3)*16
// -------------------------------------------------------------------------
__global__ __launch_bounds__(256)
void prep_w_kernel(const float* __restrict__ W, _Float16* __restrict__ wfrag)
{
    int i = blockIdx.x * 256 + threadIdx.x;          // < 221184
    int e    = i & 15;
    int lane = (i >> 4) & 31;
    int t    = (i >> 9) & 7;
    int kb   = (i >> 12) & 1;
    int k    = i >> 13;
    int col  = t * 16 + (lane & 15);
    int kk   = kb * 32 + ((lane >> 4) * 8) + (e & 7) + ((e >> 3) * 16);
    wfrag[i] = (_Float16)W[((size_t)k * CIN + kk) * COUT + col];
}

// -------------------------------------------------------------------------
// Gather + GEMM + partial BN stats, double-buffered LDS A tile.
// 256 threads = 8 waves; wave w owns rows [w*16, w*16+16) of the 128-row
// tile and all 128 output columns (8 tiles of 16x16, fp32 accumulators).
// -------------------------------------------------------------------------
__device__ __forceinline__ void gather_rows(const float* __restrict__ feats,
                                            const int* __restrict__ nidx,
                                            const unsigned char* __restrict__ nmask,
                                            int grow, int h, int k, v8h pk[4])
{
    const int   g = nidx[grow + k];
    const float m = nmask[grow + k] ? 1.0f : 0.0f;
    const float4* fp = (const float4*)(feats + (size_t)g * CIN + h * 32);
#pragma unroll
    for (int i = 0; i < 4; ++i) {
        float4 x0 = fp[2 * i];
        float4 x1 = fp[2 * i + 1];
        pk[i][0] = (_Float16)(x0.x * m); pk[i][1] = (_Float16)(x0.y * m);
        pk[i][2] = (_Float16)(x0.z * m); pk[i][3] = (_Float16)(x0.w * m);
        pk[i][4] = (_Float16)(x1.x * m); pk[i][5] = (_Float16)(x1.y * m);
        pk[i][6] = (_Float16)(x1.z * m); pk[i][7] = (_Float16)(x1.w * m);
    }
}

__global__ __launch_bounds__(256)
void conv_kernel(const float* __restrict__ feats,
                 const int* __restrict__ nidx,
                 const unsigned char* __restrict__ nmask,  // jnp.bool_ -> 1 byte
                 const _Float16* __restrict__ wfrag,
                 float* __restrict__ out,
                 float* __restrict__ bsum,
                 float* __restrict__ bsq)
{
    __shared__ _Float16 lds_a[2][TILE_M * CIN];   // 2 x 16 KB double-buffered A
    __shared__ float    lds_red[2][8][COUT];      // 8 KB reduction scratch

    const int tid  = threadIdx.x;
    const int lane = tid & 31;
    const int w    = tid >> 5;
    const int R0   = blockIdx.x * TILE_M;

    v8f acc[8];
#pragma unroll
    for (int t = 0; t < 8; ++t)
#pragma unroll
        for (int v = 0; v < 8; ++v) acc[t][v] = 0.0f;

    // gather assignment: 2 threads per row, 32 channels each
    const int r = tid >> 1;
    const int h = tid & 1;
    const int grow = (R0 + r) * KOFF;
    const int dsto = r * CIN + h * 32;

    // A-fragment addressing (ISA 16-bit A 16x32 layout)
    const int arow = w * 16 + (lane & 15);
    const int k0   = (lane >> 4) * 8;             // 0 or 8
    const int abase = arow * CIN + k0;

    v8h pk[4];                                    // in-flight gathered row (f16)

    // prologue: fill buffer 0 with offset k=0
    gather_rows(feats, nidx, nmask, grow, h, 0, pk);
    {
        _Float16* dstb = &lds_a[0][dsto];
#pragma unroll
        for (int i = 0; i < 4; ++i) *(v8h*)(dstb + i * 8) = pk[i];
    }

    for (int k = 0; k < KOFF; ++k) {
        const int cur = k & 1;

        // issue next iteration's gather loads before the barrier so the
        // random-access L2 reads overlap the WMMA stream below
        if (k + 1 < KOFF)
            gather_rows(feats, nidx, nmask, grow, h, k + 1, pk);

        __syncthreads();                          // lds_a[cur] ready

        // ---- A fragments for K = 0..31 and 32..63 ----
        const v4u* pa = (const v4u*)&lds_a[cur][abase];
        Frag16 a0, a1;
        a0.u[0] = pa[0]; a0.u[1] = pa[2];         // K: k0..k0+7 , k0+16..k0+23
        a1.u[0] = pa[4]; a1.u[1] = pa[6];         // K: 32+k0.. , 48+k0..

        // ---- 8 column tiles x 2 K-blocks of WMMA ----
        const _Float16* wk = wfrag + (size_t)k * 8192 + lane * 16;
#pragma unroll
        for (int t = 0; t < 8; ++t) {
            Frag16 b0, b1;
            const v4u* pb0 = (const v4u*)(wk + t * 512);
            const v4u* pb1 = (const v4u*)(wk + 4096 + t * 512);
            b0.u[0] = pb0[0]; b0.u[1] = pb0[1];
            b1.u[0] = pb1[0]; b1.u[1] = pb1[1];
            acc[t] = __builtin_amdgcn_wmma_f32_16x16x32_f16(
                false, a0.h, false, b0.h, (short)0, acc[t], false, false);
            acc[t] = __builtin_amdgcn_wmma_f32_16x16x32_f16(
                false, a1.h, false, b1.h, (short)0, acc[t], false, false);
        }

        // park next tile in the alternate buffer; reads of that buffer
        // finished before the barrier above, so this is race-free
        if (k + 1 < KOFF) {
            _Float16* dstb = &lds_a[1 - cur][dsto];
#pragma unroll
            for (int i = 0; i < 4; ++i) *(v8h*)(dstb + i * 8) = pk[i];
        }
    }

    // ---- store y and accumulate per-block column sums / sumsq ----
    const int colbase = lane & 15;
    const int mbase   = (lane < 16) ? 0 : 8;      // C/D layout: lanes 16-31 hold M=8..15
#pragma unroll
    for (int t = 0; t < 8; ++t) {
        const int col = t * 16 + colbase;
        float* orow = out + (size_t)(R0 + w * 16 + mbase) * COUT + col;
        float s = 0.0f, s2 = 0.0f;
#pragma unroll
        for (int v = 0; v < 8; ++v) {
            float y = acc[t][v];
            orow[(size_t)v * COUT] = y;
            s  += y;
            s2 += y * y;
        }
        s  += __shfl_down(s, 16, 32);             // fold M=8..15 half into lanes 0-15
        s2 += __shfl_down(s2, 16, 32);
        if (lane < 16) { lds_red[0][w][col] = s; lds_red[1][w][col] = s2; }
    }
    __syncthreads();
    if (tid < COUT) {
        float s = 0.0f, s2 = 0.0f;
#pragma unroll
        for (int ww = 0; ww < 8; ++ww) { s += lds_red[0][ww][tid]; s2 += lds_red[1][ww][tid]; }
        bsum[blockIdx.x * COUT + tid] = s;
        bsq [blockIdx.x * COUT + tid] = s2;
    }
}

// -------------------------------------------------------------------------
// Reduce the 1024 block partials in fixed order -> per-channel scale/bias.
// -------------------------------------------------------------------------
__global__ void bn_finalize_kernel(const float* __restrict__ bsum,
                                   const float* __restrict__ bsq,
                                   const float* __restrict__ gamma,
                                   const float* __restrict__ beta,
                                   float* __restrict__ scale,
                                   float* __restrict__ bias)
{
    const int c = threadIdx.x;                    // 128 threads
    float s = 0.0f, s2 = 0.0f;
    for (int b = 0; b < NBLK; ++b) {
        s  += bsum[b * COUT + c];
        s2 += bsq [b * COUT + c];
    }
    const float inv_n = 1.0f / (float)NPTS;
    const float mean  = s * inv_n;
    const float var   = s2 * inv_n - mean * mean;
    const float rstd  = rsqrtf(var + BN_EPS);
    const float g     = gamma[c];
    scale[c] = rstd * g;
    bias[c]  = beta[c] - mean * rstd * g;
}

// -------------------------------------------------------------------------
// In-place affine + LeakyReLU, one float4 per thread.
// -------------------------------------------------------------------------
__global__ __launch_bounds__(256)
void bn_apply_kernel(float* __restrict__ y,
                     const float* __restrict__ scale,
                     const float* __restrict__ bias)
{
    const int i   = blockIdx.x * 256 + threadIdx.x;   // float4 index
    const int col = (i * 4) & (COUT - 1);             // multiple of 4
    float4 v  = ((const float4*)y)[i];
    float4 sc = *(const float4*)(scale + col);
    float4 bi = *(const float4*)(bias + col);
    v.x = v.x * sc.x + bi.x;  v.y = v.y * sc.y + bi.y;
    v.z = v.z * sc.z + bi.z;  v.w = v.w * sc.w + bi.w;
    v.x = v.x > 0.0f ? v.x : NEG_SLOPE * v.x;
    v.y = v.y > 0.0f ? v.y : NEG_SLOPE * v.y;
    v.z = v.z > 0.0f ? v.z : NEG_SLOPE * v.z;
    v.w = v.w > 0.0f ? v.w : NEG_SLOPE * v.w;
    ((float4*)y)[i] = v;
}

extern "C" void kernel_launch(void* const* d_in, const int* in_sizes, int n_in,
                              void* d_out, int out_size, void* d_ws, size_t ws_size,
                              hipStream_t stream)
{
    const float*         feats = (const float*)d_in[0];
    const int*           nidx  = (const int*)d_in[1];
    const unsigned char* nmask = (const unsigned char*)d_in[2]; // jnp.bool_
    const float*         W     = (const float*)d_in[3];
    const float*         gamma = (const float*)d_in[4];
    const float*         beta  = (const float*)d_in[5];
    float* out = (float*)d_out;

    char* ws = (char*)d_ws;
    _Float16* wfrag = (_Float16*)(ws + OFF_WFRAG);
    float*    bsum  = (float*)(ws + OFF_BSUM);
    float*    bsq   = (float*)(ws + OFF_BSQ);
    float*    scale = (float*)(ws + OFF_SCALE);
    float*    bias  = (float*)(ws + OFF_BIAS);

    prep_w_kernel<<<WFRAG_ELEMS / 256, 256, 0, stream>>>(W, wfrag);
    conv_kernel<<<NBLK, 256, 0, stream>>>(feats, nidx, nmask, wfrag, out, bsum, bsq);
    bn_finalize_kernel<<<1, COUT, 0, stream>>>(bsum, bsq, gamma, beta, scale, bias);
    bn_apply_kernel<<<(NPTS * COUT / 4) / 256, 256, 0, stream>>>(out, scale, bias);
}
